// ConditionalRandomField_24550033064060
// MI455X (gfx1250) — matrix operand order repaired and checked by reading, not compiled, not run on hardware
//
#include <hip/hip_runtime.h>
#include <hip/hip_bf16.h>
#include <stdint.h>

// ---------------------------------------------------------------------------
// CRF loss for MI455X (gfx1250).
//
//   log_den:  fv' = lse_j(fv_j + T_ij) - max_j(fv_j + T_ij)   (emit cancels)
//   exp-domain:  g'[i] = (sum_j E_ij g_j) / (max_j E_ij g_j),  E = exp(T),
//                row START of E := 1 (constant row factor cancels in ratio).
//   g is kept scaled by 1/4 so every f16 product E_ij*g_j < 2^16.
//
//   sum term  -> v_wmma_f32_16x16x32_f16 (f32 accumulate)
//   max term  -> v_pk_mul_f16 + v_pk_max_num_f16 (co-executes with XDL WMMA)
//   E streamed L2->LDS with TDM (tensor_load_to_lds), double buffered,
//   synchronized with s_wait_tensorcnt + workgroup barriers.
// ---------------------------------------------------------------------------

#define NUM_TAGS  512
#define START_TAG 510
#define STOP_TAG  511
#define BATCH      64
#define SEQ_LEN   512

typedef _Float16 h2   __attribute__((ext_vector_type(2)));
typedef _Float16 h4   __attribute__((ext_vector_type(4)));
typedef _Float16 h8   __attribute__((ext_vector_type(8)));
typedef _Float16 v16h __attribute__((ext_vector_type(16)));
typedef float    v8f  __attribute__((ext_vector_type(8)));
typedef unsigned int u32x4 __attribute__((ext_vector_type(4)));
typedef int          i32x4 __attribute__((ext_vector_type(4)));
typedef int          i32x8 __attribute__((ext_vector_type(8)));

// --------------------------------------------------------------------------
// E[i][j] = exp(trans[i][j]); row START := 1.0 (see header comment).
// exp(-10000) underflows to +0 which correctly kills column STOP.
// --------------------------------------------------------------------------
__global__ __launch_bounds__(256) void crf_prep_exp(const float* __restrict__ trans,
                                                    _Float16* __restrict__ E) {
  int idx = blockIdx.x * 256 + threadIdx.x;       // 0 .. 512*512-1
  int i = idx >> 9;
  float v = (i == START_TAG) ? 1.0f : expf(trans[idx]);
  E[idx] = (_Float16)v;
}

// Low 32 bits of a flat __shared__ pointer are the LDS byte offset
// (ISA 10.2 aperture mapping).
__device__ __forceinline__ unsigned lds_off(const void* p) {
  return (unsigned)(size_t)p;
}

// TDM: load a 512-row x 32-col f16 tile (row stride 512 elems) into LDS.
// D# per CDNA5 ISA; groups 2/3 zero (tile_dim3=0).
__device__ __forceinline__ void tdm_load_tile(const _Float16* gtile, unsigned ldsaddr) {
  unsigned long long ga = (unsigned long long)gtile;
  u32x4 g0;
  g0.x = 1u;                                               // count = 1
  g0.y = ldsaddr;                                          // lds_addr
  g0.z = (unsigned)ga;                                     // global_addr[31:0]
  g0.w = (unsigned)((ga >> 32) & 0x01FFFFFFu) | (2u << 30);// addr[56:32] | type=2
  i32x8 g1 = { (int)(1u << 16),        // data_size = 1 (2-byte elements)
               (int)(0x0200u << 16),   // tensor_dim0 = 512   (bits 79:48 lo)
               (int)(0x0200u << 16),   // tensor_dim1 = 512   (bits 111:80 lo)
               (int)(32u << 16),       // tile_dim0   = 32    (bits 127:112)
               0x0200,                 // tile_dim1   = 512, tile_dim2 = 0
               512,                    // tensor_dim0_stride = 512
               0, 0 };                 // tensor_dim1_stride = 0
  i32x4 z4 = { 0, 0, 0, 0 };
#if __clang_major__ >= 23
  i32x8 z8 = { 0, 0, 0, 0, 0, 0, 0, 0 };
  __builtin_amdgcn_tensor_load_to_lds(g0, g1, z4, z4, z8, 0);
#else
  __builtin_amdgcn_tensor_load_to_lds(g0, g1, z4, z4, 0);
#endif
}

// --------------------------------------------------------------------------
// Forward recursion. 16 workgroups x 4 batches; 512 threads (16 waves).
// Wave w owns rows 32w..32w+31: two 16x16 WMMA M-tiles + one max-plus row/lane.
// N-tile: 16 cols = 4 batches x 4 replicas (only cols 0..3 written back).
// --------------------------------------------------------------------------
__global__ __launch_bounds__(512, 1) void crf_forward(
    const _Float16* __restrict__ E,    // [512][512] exp(transitions)
    const float*    __restrict__ trans,// [512][512] (terminal row only)
    float*          __restrict__ den)  // [64]
{
  __shared__ __align__(16) _Float16 Etile[2][NUM_TAGS * 32]; // 2 x 32 KB
  __shared__ __align__(16) _Float16 G[2][NUM_TAGS][4];       // g = exp(fv)/4
  __shared__ float Ssum[NUM_TAGS][4];
  __shared__ float Mmax[NUM_TAGS][4];

  const int tid  = threadIdx.x;
  const int lane = tid & 31;
  const int wave = tid >> 5;            // 0..15
  const int hi   = (lane >= 16) ? 1 : 0;

  for (int j = tid; j < NUM_TAGS; j += 512) {
    float g0 = (j == START_TAG) ? 0.25f : 0.0f;
#pragma unroll
    for (int b = 0; b < 4; ++b) G[0][j][b] = (_Float16)g0;
  }
  if (wave == 0) tdm_load_tile(E, lds_off(&Etile[0][0]));    // chunk 0 in flight
  __syncthreads();

  int cur = 0;
  for (int t = 0; t < SEQ_LEN; ++t) {
    v8f acc0 = {}; v8f acc1 = {};
    h2 zero2 = { (_Float16)0.f, (_Float16)0.f };
    h2 mac[4] = { zero2, zero2, zero2, zero2 };              // products are >= 0
    const _Float16* gcur = &G[cur][0][0];

    for (int c = 0; c < 16; ++c) {                           // K chunks of 32
      const _Float16* tile = &Etile[c & 1][0];
      if (wave == 0) {
        if (c < 15) {
          tdm_load_tile(E + (c + 1) * 32, lds_off(&Etile[(c + 1) & 1][0]));
          __builtin_amdgcn_s_wait_tensorcnt(1);              // oldest (c) done
        } else {
          __builtin_amdgcn_s_wait_tensorcnt(0);
        }
      }
      __syncthreads();                                       // tile[c] visible

      // ---- B fragment: B[k,n] = g[c*32 + k][n&3],  k = h + 16*hi ----
      v16h bf;
      {
        int jb = c * 32 + hi * 16;
#pragma unroll
        for (int h = 0; h < 16; ++h) bf[h] = gcur[(jb + h) * 4 + (lane & 3)];
      }
      // ---- A fragments: two ds_load_b128 each, contiguous k = 16*hi + s, same as B ----
      v16h a0, a1;
      {
        int r0 = 32 * wave + (lane & 15);
        int q  = 2 * hi;                                      // k = 16*hi + s
        const h8* p = (const h8*)(tile + r0 * 32);
        ((h8*)&a0)[0] = p[q];       ((h8*)&a0)[1] = p[1 + q];
        const h8* r = (const h8*)(tile + (r0 + 16) * 32);
        ((h8*)&a1)[0] = r[q];       ((h8*)&a1)[1] = r[1 + q];
      }
      acc0 = __builtin_amdgcn_wmma_f32_16x16x32_f16(false, a0, false, bf,
                                                    (short)0, acc0, false, false);
      acc1 = __builtin_amdgcn_wmma_f32_16x16x32_f16(false, a1, false, bf,
                                                    (short)0, acc1, false, false);

      // ---- max-plus: packed f16 mul+max over this chunk's 32 j's ----
      {
        const h8* erow = (const h8*)(tile + (32 * wave + lane) * 32);
        const h4* grow = (const h4*)gcur;
#pragma unroll
        for (int q8 = 0; q8 < 4; ++q8) {
          h8 e8 = erow[q8];
#pragma unroll
          for (int p2 = 0; p2 < 4; ++p2) {
            int j0 = c * 32 + q8 * 8 + p2 * 2;
            h4 ga = grow[j0];
            h4 gb = grow[j0 + 1];
            h2 e2 = { e8[p2 * 2], e8[p2 * 2 + 1] };
#pragma unroll
            for (int b = 0; b < 4; ++b) {
              h2 g2 = { ga[b], gb[b] };
              mac[b] = __builtin_elementwise_max(mac[b], e2 * g2);
            }
          }
        }
      }
      __syncthreads();                    // all done with tile[c] before reuse
    }

    // ---- spill S (WMMA D layout: n = lane%16, M = v + 8*hi) and M ----
    {
      int n = lane & 15;
      if (n < 4) {
        int r = 32 * wave + hi * 8;
#pragma unroll
        for (int v = 0; v < 8; ++v) {
          Ssum[r + v][n]      = acc0[v];
          Ssum[r + 16 + v][n] = acc1[v];
        }
      }
      int i = 32 * wave + lane;
#pragma unroll
      for (int b = 0; b < 4; ++b)
        Mmax[i][b] = fmaxf((float)mac[b][0], (float)mac[b][1]);
    }
    __syncthreads();
    // ---- g_{t+1} = 0.25 * S / M  (the 1/4 scale cancels in every ratio) ----
    {
      int i = 32 * wave + lane;
      int nxt = cur ^ 1;
#pragma unroll
      for (int b = 0; b < 4; ++b)
        G[nxt][i][b] = (_Float16)(0.25f * Ssum[i][b] / Mmax[i][b]);
    }
    if (wave == 0 && t + 1 < SEQ_LEN)            // hide next step's TDM latency
      tdm_load_tile(E, lds_off(&Etile[0][0]));
    cur ^= 1;
  }

  __syncthreads();
  // terminal: log_den = log(sum_j e^{T[STOP,j]} g_j) - log(max_j e^{T[STOP,j]} g_j)
  if (wave == 0) {
#pragma unroll
    for (int b = 0; b < 4; ++b) {
      float s = 0.0f, m = 0.0f;
      for (int j = lane; j < NUM_TAGS; j += 32) {
        float v = expf(trans[STOP_TAG * NUM_TAGS + j]) * (float)G[cur][j][b];
        s += v;
        m = fmaxf(m, v);
      }
#pragma unroll
      for (int o = 16; o > 0; o >>= 1) {
        s += __shfl_xor(s, o, 32);
        m = fmaxf(m, __shfl_xor(m, o, 32));
      }
      if (lane == 0) den[blockIdx.x * 4 + b] = logf(s) - logf(m);
    }
  }
}

// --------------------------------------------------------------------------
// Numerator: boundary terms + sum_t ( T[tag_t, tag_{t-1}] + inp[b,t-1,tag_t] ).
// Fixed-order tree reduction => deterministic.
// --------------------------------------------------------------------------
__global__ __launch_bounds__(256) void crf_numerator(
    const float* __restrict__ inputs, const int* __restrict__ tags,
    const float* __restrict__ trans, float* __restrict__ num)
{
  __shared__ float red[256];
  int b = blockIdx.x, tid = threadIdx.x;
  const int* tg = tags + b * SEQ_LEN;
  float acc = 0.0f;
  for (int t = 1 + tid; t < SEQ_LEN; t += 256) {
    int ct = tg[t], pt = tg[t - 1];
    acc += trans[ct * NUM_TAGS + pt] +
           inputs[((size_t)b * SEQ_LEN + (t - 1)) * NUM_TAGS + ct];
  }
  red[tid] = acc;
  __syncthreads();
  for (int s = 128; s > 0; s >>= 1) {
    if (tid < s) red[tid] += red[tid + s];
    __syncthreads();
  }
  if (tid == 0)
    num[b] = red[0] + trans[tg[0] * NUM_TAGS + START_TAG] +
             trans[STOP_TAG * NUM_TAGS + tg[SEQ_LEN - 1]];
}

__global__ void crf_finalize(const float* __restrict__ num,
                             const float* __restrict__ den,
                             float* __restrict__ out) {
  if (threadIdx.x == 0 && blockIdx.x == 0) {
    float s = 0.0f;
    for (int b = 0; b < BATCH; ++b) s += num[b] - den[b];
    out[0] = s;
  }
}

extern "C" void kernel_launch(void* const* d_in, const int* in_sizes, int n_in,
                              void* d_out, int out_size, void* d_ws, size_t ws_size,
                              hipStream_t stream) {
  const float* inputs = (const float*)d_in[0];   // [64][512][512] f32
  const int*   tags   = (const int*)d_in[1];     // [64][512] i32
  const float* trans  = (const float*)d_in[2];   // [512][512] f32

  _Float16* E   = (_Float16*)d_ws;                                   // 512 KB
  float*    den = (float*)((char*)d_ws + NUM_TAGS * NUM_TAGS * sizeof(_Float16));
  float*    num = den + BATCH;

  crf_prep_exp <<<(NUM_TAGS * NUM_TAGS) / 256, 256, 0, stream>>>(trans, E);
  crf_forward  <<<BATCH / 4, 512, 0, stream>>>(E, trans, den);
  crf_numerator<<<BATCH, 256, 0, stream>>>(inputs, tags, trans, num);
  crf_finalize <<<1, 64, 0, stream>>>(num, den, (float*)d_out);

  (void)in_sizes; (void)n_in; (void)out_size; (void)ws_size;
}
